// FullLanguageZone_90314572300855
// MI455X (gfx1250) — matrix operand order repaired and checked by reading, not compiled, not run on hardware
//
#include <hip/hip_runtime.h>
#include <hip/hip_bf16.h>

// ---------------- problem constants (match reference) ----------------
#define NTOK   8192      // B*S
#define DMODEL 1024
#define HDIM   2048
#define MMOE   64
#define EEXP   8
#define HHALF  1024      // H/2

typedef __attribute__((ext_vector_type(16))) __bf16 v16bf;
typedef __attribute__((ext_vector_type(8)))  float  v8f;

__device__ __forceinline__ unsigned short f2bf(float f) {
  union { float f; unsigned u; } v; v.f = f;
  unsigned r = v.u + 0x7FFFu + ((v.u >> 16) & 1u);   // round-to-nearest-even
  return (unsigned short)(r >> 16);
}
__device__ __forceinline__ float sigmoidf_(float x) { return 1.f / (1.f + __expf(-x)); }

// ---- CDNA5 async global->LDS copy (ASYNCcnt-tracked), 16B per lane ----
__device__ __forceinline__ void async_copy16(const unsigned short* gptr, unsigned short* lptr) {
  unsigned lds = (unsigned)(unsigned long long)lptr;   // low 32 bits of LDS generic addr = LDS offset
  asm volatile("global_load_async_to_lds_b128 %0, %1, off" :: "v"(lds), "v"(gptr) : "memory");
}
__device__ __forceinline__ void wait_async() {
  asm volatile("s_wait_asynccnt 0" ::: "memory");
}

union FragU { v16bf v; uint4 q[2]; };

// A fragment (16x32 bf16, M x K): lane holds row m; element j -> K = 16*(j>>3) + 8*half + (j&7)
__device__ __forceinline__ v16bf load_frag_a(const unsigned short* lds, int row, int stride,
                                             int kbase, int half) {
  FragU u;
  const unsigned short* p = lds + row * stride + kbase + 8 * half;
  u.q[0] = *(const uint4*)p;
  u.q[1] = *(const uint4*)(p + 16);
  return u.v;
}
// B fragment (32x16 bf16, K x N) from K-contiguous transposed layout: element j -> K = 16*(j>>3) + 8*half + (j&7)
__device__ __forceinline__ v16bf load_frag_b(const unsigned short* ldsT, int col, int stride,
                                             int kbase, int half) {
  FragU u;
  const unsigned short* p = ldsT + col * stride + kbase + 8 * half;
  u.q[0] = *(const uint4*)p;
  u.q[1] = *(const uint4*)(p + 16);
  return u.v;
}

// ---------------- fp32 -> bf16 conversion (activations) ----------------
__global__ __launch_bounds__(256) void f32_to_bf16_kernel(const float* __restrict__ src,
                                                          unsigned short* __restrict__ dst, int n) {
  int i = blockIdx.x * 256 + threadIdx.x;
  if (i < n) dst[i] = f2bf(src[i]);
}

// ---------------- fp32 [R,C] -> bf16 transposed [C,R] (weights, batched) ----------------
__global__ __launch_bounds__(256) void transpose_cvt_kernel(const float* __restrict__ src,
                                                            unsigned short* __restrict__ dst,
                                                            int R, int C) {
  __shared__ float s[64][65];
  const int r0 = blockIdx.y * 64, c0 = blockIdx.x * 64;
  const size_t base = (size_t)blockIdx.z * R * C;
  const int tid = threadIdx.x;
#pragma unroll
  for (int i = 0; i < 16; ++i) {
    int idx = tid + 256 * i;
    int r = idx >> 6, c = idx & 63;
    s[r][c] = src[base + (size_t)(r0 + r) * C + c0 + c];
  }
  __syncthreads();
#pragma unroll
  for (int i = 0; i < 16; ++i) {
    int idx = tid + 256 * i;
    int c = idx >> 6, r = idx & 63;
    dst[base + (size_t)(c0 + c) * R + r0 + r] = f2bf(s[r][c]);
  }
}

// ---------------- big WMMA GEMM: out = act(A[M,K] @ Wt[N,K]^T + bias) ----------------
// 256 threads (8 waves); block tile 128(M) x 128(N); K step 32; async double-buffered LDS.
__global__ __launch_bounds__(256) void gemm_bf16_big_kernel(
    const unsigned short* __restrict__ A, const unsigned short* __restrict__ Wt,
    const float* __restrict__ bias, float* __restrict__ outF, unsigned short* __restrict__ outB,
    int M, int N, int K, int act) {
  __shared__ __align__(16) unsigned short sA[2][128 * 32];
  __shared__ __align__(16) unsigned short sB[2][128 * 32];
  const int tid  = threadIdx.x;
  const int w    = tid >> 5;
  const int lane = tid & 31;
  const int ln   = lane & 15;
  const int half = lane >> 4;
  const int wm   = w & 3;    // 4 M-strips of 32
  const int wn   = w >> 2;   // 2 N-strips of 64
  const int m0 = blockIdx.y * 128;
  const int n0 = blockIdx.x * 128;

  v8f acc[8];
#pragma unroll
  for (int t = 0; t < 8; ++t) acc[t] = (v8f){0.f, 0.f, 0.f, 0.f, 0.f, 0.f, 0.f, 0.f};

  // issue first stage (buffer 0)
  {
#pragma unroll
    for (int i = 0; i < 2; ++i) {
      int c = tid + 256 * i;
      int row = c >> 2, ko = (c & 3) << 3;
      async_copy16(A + (size_t)(m0 + row) * K + ko, &sA[0][row * 32 + ko]);
      async_copy16(Wt + (size_t)(n0 + row) * K + ko, &sB[0][row * 32 + ko]);
    }
  }

  int buf = 0;
  for (int kt = 0; kt < K; kt += 32) {
    if (kt + 64 < K)  // long-range hint -> global_prefetch_b8
      __builtin_prefetch(A + (size_t)(m0 + (tid >> 1)) * K + kt + 64, 0, 0);
    wait_async();
    __syncthreads();
    if (kt + 32 < K) {  // issue next stage into other buffer while computing this one
#pragma unroll
      for (int i = 0; i < 2; ++i) {
        int c = tid + 256 * i;
        int row = c >> 2, ko = (c & 3) << 3;
        async_copy16(A + (size_t)(m0 + row) * K + kt + 32 + ko, &sA[buf ^ 1][row * 32 + ko]);
        async_copy16(Wt + (size_t)(n0 + row) * K + kt + 32 + ko, &sB[buf ^ 1][row * 32 + ko]);
      }
    }
    const unsigned short* cA = sA[buf];
    const unsigned short* cB = sB[buf];
    v16bf a0 = load_frag_a(cA, wm * 32 + ln, 32, 0, half);
    v16bf a1 = load_frag_a(cA, wm * 32 + 16 + ln, 32, 0, half);
#pragma unroll
    for (int t = 0; t < 4; ++t) {
      v16bf b = load_frag_b(cB, wn * 64 + t * 16 + ln, 32, 0, half);
      acc[t]     = __builtin_amdgcn_wmma_f32_16x16x32_bf16(false, a0, false, b, (short)0, acc[t],
                                                           false, false);
      acc[4 + t] = __builtin_amdgcn_wmma_f32_16x16x32_bf16(false, a1, false, b, (short)0,
                                                           acc[4 + t], false, false);
    }
    buf ^= 1;
  }
  // epilogue: C/D layout -> (row = i + 8*half, col = lane%16)
#pragma unroll
  for (int s = 0; s < 2; ++s) {
#pragma unroll
    for (int t = 0; t < 4; ++t) {
      int gn = n0 + wn * 64 + t * 16 + ln;
      float bv = bias ? bias[gn] : 0.f;
#pragma unroll
      for (int i = 0; i < 8; ++i) {
        int gm = m0 + wm * 32 + s * 16 + i + 8 * half;
        float v = acc[s * 4 + t][i] + bv;
        if (act) v = sigmoidf_(v);
        if (outF) outF[(size_t)gm * N + gn] = v;
        if (outB) outB[(size_t)gm * N + gn] = f2bf(v);
      }
    }
  }
}

// ---------------- small WMMA GEMM (N==64): out = act(A[M,K] @ Wt[64,K]^T + bias) ----------------
// 128 threads (4 waves); block tile 64 x 64; async single-buffered staging.
__global__ __launch_bounds__(128) void gemm_bf16_n64_kernel(
    const unsigned short* __restrict__ A, const unsigned short* __restrict__ Wt,
    const float* __restrict__ bias, float* __restrict__ outF, unsigned short* __restrict__ outB,
    int M, int K, int act) {
  const int N = 64;
  __shared__ __align__(16) unsigned short sA[64 * 32];
  __shared__ __align__(16) unsigned short sB[64 * 32];
  const int tid  = threadIdx.x;
  const int w    = tid >> 5;
  const int lane = tid & 31;
  const int ln   = lane & 15;
  const int half = lane >> 4;
  const int m0 = blockIdx.y * 64;

  v8f acc[4];
#pragma unroll
  for (int t = 0; t < 4; ++t) acc[t] = (v8f){0.f, 0.f, 0.f, 0.f, 0.f, 0.f, 0.f, 0.f};

  for (int kt = 0; kt < K; kt += 32) {
#pragma unroll
    for (int i = 0; i < 2; ++i) {
      int c = tid + 128 * i;
      int row = c >> 2, ko = (c & 3) << 3;
      async_copy16(A + (size_t)(m0 + row) * K + kt + ko, &sA[row * 32 + ko]);
      async_copy16(Wt + (size_t)row * K + kt + ko, &sB[row * 32 + ko]);
    }
    wait_async();
    __syncthreads();
    v16bf a = load_frag_a(sA, w * 16 + ln, 32, 0, half);
#pragma unroll
    for (int t = 0; t < 4; ++t) {
      v16bf b = load_frag_b(sB, t * 16 + ln, 32, 0, half);
      acc[t] = __builtin_amdgcn_wmma_f32_16x16x32_bf16(false, a, false, b, (short)0, acc[t],
                                                       false, false);
    }
    __syncthreads();
  }
#pragma unroll
  for (int t = 0; t < 4; ++t) {
    int gn = t * 16 + ln;
    float bv = bias ? bias[gn] : 0.f;
#pragma unroll
    for (int i = 0; i < 8; ++i) {
      int gm = m0 + w * 16 + i + 8 * half;
      float v = acc[t][i] + bv;
      if (act) v = sigmoidf_(v);
      if (outF) outF[(size_t)gm * N + gn] = v;
      if (outB) outB[(size_t)gm * N + gn] = f2bf(v);
    }
  }
}

// ---------------- router: logits = tanh(cont@rW1+b1)@rW2+b2; softmax; top-2 renorm ----------------
__global__ __launch_bounds__(256) void router_kernel(
    const float* __restrict__ contF, const float* __restrict__ rW1, const float* __restrict__ rb1,
    const float* __restrict__ rW2, const float* __restrict__ rb2, float* __restrict__ gates) {
  __shared__ float sW1[64 * 64];
  __shared__ float sB1[64];
  __shared__ float sW2[64 * 8];
  __shared__ float sB2[8];
  int tid = threadIdx.x;
  for (int i = tid; i < 64 * 64; i += 256) sW1[i] = rW1[i];
  for (int i = tid; i < 64 * 8; i += 256) sW2[i] = rW2[i];
  if (tid < 64) sB1[tid] = rb1[tid];
  if (tid < 8) sB2[tid] = rb2[tid];
  __syncthreads();

  int tok = blockIdx.x * 256 + tid;
  float c[64];
#pragma unroll
  for (int k = 0; k < 64; ++k) c[k] = contF[(size_t)tok * 64 + k];
  float logit[8];
#pragma unroll
  for (int e = 0; e < 8; ++e) logit[e] = sB2[e];
  for (int j = 0; j < 64; ++j) {
    float s = sB1[j];
#pragma unroll
    for (int k = 0; k < 64; ++k) s += c[k] * sW1[k * 64 + j];
    float t = tanhf(s);
#pragma unroll
    for (int e = 0; e < 8; ++e) logit[e] += t * sW2[j * 8 + e];
  }
  float mx = logit[0];
#pragma unroll
  for (int e = 1; e < 8; ++e) mx = fmaxf(mx, logit[e]);
  float p[8];
#pragma unroll
  for (int e = 0; e < 8; ++e) p[e] = __expf(logit[e] - mx);
  int i1 = 0; float m1 = p[0];
#pragma unroll
  for (int e = 1; e < 8; ++e) if (p[e] > m1) { m1 = p[e]; i1 = e; }
  int i2 = -1; float m2 = -1.f;
#pragma unroll
  for (int e = 0; e < 8; ++e) if (e != i1 && p[e] > m2) { m2 = p[e]; i2 = e; }
  float inv = 1.f / (m1 + m2);
#pragma unroll
  for (int e = 0; e < 8; ++e) {
    float g = (e == i1) ? m1 * inv : ((e == i2) ? m2 * inv : 0.f);
    gates[(size_t)tok * 8 + e] = g;
  }
}

// ---------------- fused dense-MoE expert kernel ----------------
// out[n,:64] = sum_e gate[n,e] * ( relu(cont[n,:]@We1[e] + be1[e]) @ We2[e] + be2[e] )
// Uses pre-transposed We1t[e][h][m] (1024x64) and We2t[e][mcol][hh] (64x1024).
__global__ __launch_bounds__(128) void expert_kernel(
    const unsigned short* __restrict__ contBF, const float* __restrict__ gates,
    const unsigned short* __restrict__ We1t, const float* __restrict__ be1,
    const unsigned short* __restrict__ We2t, const float* __restrict__ be2,
    float* __restrict__ outF, unsigned short* __restrict__ outB) {
  __shared__ __align__(16) unsigned short sC[64 * 64];    // cont tile   [token][k]
  __shared__ __align__(16) unsigned short sW1t[64 * 64];  // We1t chunk  [h][m]
  __shared__ __align__(16) unsigned short sW2t[64 * 64];  // We2t chunk  [mcol][h]
  __shared__ __align__(16) unsigned short sH[64 * 64];    // relu(h)     [token][h]
  const int tid  = threadIdx.x;
  const int w    = tid >> 5;
  const int lane = tid & 31;
  const int ln   = lane & 15;
  const int half = lane >> 4;
  const int tok0 = blockIdx.x * 64;

  // stage cont tile once (contiguous async copies)
#pragma unroll
  for (int i = 0; i < 4; ++i) {
    int c = tid + 128 * i;
    async_copy16(contBF + (size_t)tok0 * 64 + c * 8, &sC[c * 8]);
  }

  v8f outacc[4];
#pragma unroll
  for (int t = 0; t < 4; ++t) outacc[t] = (v8f){0.f, 0.f, 0.f, 0.f, 0.f, 0.f, 0.f, 0.f};

  for (int e = 0; e < EEXP; ++e) {
    v8f eacc[4];
#pragma unroll
    for (int t = 0; t < 4; ++t) eacc[t] = (v8f){0.f, 0.f, 0.f, 0.f, 0.f, 0.f, 0.f, 0.f};

    for (int cc = 0; cc < HHALF / 64; ++cc) {
      __syncthreads();  // protect previous chunk's sW1t/sW2t reads before restage
      // We1t chunk: rows h in [cc*64, cc*64+64), 64 cols -> one contiguous 8KB block
      {
        const unsigned short* src1 = We1t + (size_t)e * HHALF * 64 + (size_t)(cc * 64) * 64;
        const unsigned short* src2 = We2t + (size_t)e * 64 * HHALF + cc * 64;
#pragma unroll
        for (int i = 0; i < 4; ++i) {
          int c = tid + 128 * i;
          async_copy16(src1 + c * 8, &sW1t[c * 8]);
          int row = c >> 3, ko = (c & 7) << 3;  // We2t rows stride HHALF
          async_copy16(src2 + (size_t)row * HHALF + ko, &sW2t[row * 64 + ko]);
        }
      }
      wait_async();
      __syncthreads();
      // GEMM1: h = cont(64x64) @ We1chunk(64x64)
      v8f hacc[4];
#pragma unroll
      for (int t = 0; t < 4; ++t) hacc[t] = (v8f){0.f, 0.f, 0.f, 0.f, 0.f, 0.f, 0.f, 0.f};
#pragma unroll
      for (int ks = 0; ks < 64; ks += 32) {
        v16bf a = load_frag_a(sC, w * 16 + ln, 64, ks, half);
#pragma unroll
        for (int t = 0; t < 4; ++t) {
          v16bf b = load_frag_b(sW1t, t * 16 + ln, 64, ks, half);
          hacc[t] = __builtin_amdgcn_wmma_f32_16x16x32_bf16(false, a, false, b, (short)0, hacc[t],
                                                            false, false);
        }
      }
      // relu(+be1) -> sH (bf16)
#pragma unroll
      for (int t = 0; t < 4; ++t) {
        int col = t * 16 + ln;
        float bv = be1[(size_t)e * HHALF + cc * 64 + col];
#pragma unroll
        for (int i = 0; i < 8; ++i) {
          int row = w * 16 + i + 8 * half;
          sH[row * 64 + col] = f2bf(fmaxf(hacc[t][i] + bv, 0.f));
        }
      }
      __syncthreads();
      // GEMM2: eacc += h(64x64) @ We2chunk(64x64)
#pragma unroll
      for (int ks = 0; ks < 64; ks += 32) {
        v16bf a = load_frag_a(sH, w * 16 + ln, 64, ks, half);
#pragma unroll
        for (int t = 0; t < 4; ++t) {
          v16bf b = load_frag_b(sW2t, t * 16 + ln, 64, ks, half);
          eacc[t] = __builtin_amdgcn_wmma_f32_16x16x32_bf16(false, a, false, b, (short)0, eacc[t],
                                                            false, false);
        }
      }
    }
    // gate-weighted combine (+be2)
    float gv[8];
#pragma unroll
    for (int i = 0; i < 8; ++i)
      gv[i] = gates[(size_t)(tok0 + w * 16 + i + 8 * half) * EEXP + e];
#pragma unroll
    for (int t = 0; t < 4; ++t) {
      int col = t * 16 + ln;
      float b2 = be2[(size_t)e * 64 + col];
#pragma unroll
      for (int i = 0; i < 8; ++i) outacc[t][i] += gv[i] * (eacc[t][i] + b2);
    }
  }
  // write out_flat (fp32 + bf16)
#pragma unroll
  for (int t = 0; t < 4; ++t) {
    int col = t * 16 + ln;
#pragma unroll
    for (int i = 0; i < 8; ++i) {
      int row = tok0 + w * 16 + i + 8 * half;
      float v = outacc[t][i];
      outF[(size_t)row * 64 + col] = v;
      outB[(size_t)row * 64 + col] = f2bf(v);
    }
  }
}

// ---------------- in-place LayerNorm over D=1024 ----------------
__global__ __launch_bounds__(256) void ln_kernel(float* __restrict__ out,
                                                 const float* __restrict__ g,
                                                 const float* __restrict__ b) {
  __shared__ float red[256];
  const int row = blockIdx.x, tid = threadIdx.x;
  float x[4];
  float s = 0.f;
#pragma unroll
  for (int i = 0; i < 4; ++i) {
    x[i] = out[(size_t)row * 1024 + tid + i * 256];
    s += x[i];
  }
  red[tid] = s;
  __syncthreads();
  for (int off = 128; off > 0; off >>= 1) {
    if (tid < off) red[tid] += red[tid + off];
    __syncthreads();
  }
  float mu = red[0] * (1.f / 1024.f);
  __syncthreads();
  float s2 = 0.f;
#pragma unroll
  for (int i = 0; i < 4; ++i) {
    float d = x[i] - mu;
    s2 += d * d;
  }
  red[tid] = s2;
  __syncthreads();
  for (int off = 128; off > 0; off >>= 1) {
    if (tid < off) red[tid] += red[tid + off];
    __syncthreads();
  }
  float rs = rsqrtf(red[0] * (1.f / 1024.f) + 1e-5f);
#pragma unroll
  for (int i = 0; i < 4; ++i) {
    int col = tid + i * 256;
    out[(size_t)row * 1024 + col] = (x[i] - mu) * rs * g[col] + b[col];
  }
}

// ---------------- host orchestration ----------------
extern "C" void kernel_launch(void* const* d_in, const int* in_sizes, int n_in,
                              void* d_out, int out_size, void* d_ws, size_t ws_size,
                              hipStream_t stream) {
  const float* X     = (const float*)d_in[0];
  const float* enc_W = (const float*)d_in[1];
  const float* enc_b = (const float*)d_in[2];
  const float* s2c_W = (const float*)d_in[3];
  const float* s2c_b = (const float*)d_in[4];
  const float* rW1   = (const float*)d_in[5];
  const float* rb1   = (const float*)d_in[6];
  const float* rW2   = (const float*)d_in[7];
  const float* rb2   = (const float*)d_in[8];
  const float* We1   = (const float*)d_in[9];
  const float* be1   = (const float*)d_in[10];
  const float* We2   = (const float*)d_in[11];
  const float* be2   = (const float*)d_in[12];
  const float* c2s_W = (const float*)d_in[13];
  const float* c2s_b = (const float*)d_in[14];
  const float* dec_W = (const float*)d_in[15];
  const float* dec_b = (const float*)d_in[16];
  const float* ln_g  = (const float*)d_in[17];
  const float* ln_b  = (const float*)d_in[18];
  float* out = (float*)d_out;

  char* ws = (char*)d_ws;
  size_t o = 0;
  auto alloc = [&](size_t bytes) { size_t r = o; o += (bytes + 255) & ~(size_t)255; return r; };
  unsigned short* Xbf   = (unsigned short*)(ws + alloc((size_t)NTOK * DMODEL * 2));
  unsigned short* encWt = (unsigned short*)(ws + alloc((size_t)DMODEL * HDIM * 2));   // [H][D]
  unsigned short* spkBF = (unsigned short*)(ws + alloc((size_t)NTOK * HDIM * 2));     // reused
  unsigned short* s2cWt = (unsigned short*)(ws + alloc((size_t)HDIM * MMOE * 2));     // [64][2048]
  float*          contF = (float*)(ws + alloc((size_t)NTOK * MMOE * 4));
  unsigned short* contB = (unsigned short*)(ws + alloc((size_t)NTOK * MMOE * 2));
  float*          gatesP= (float*)(ws + alloc((size_t)NTOK * EEXP * 4));
  unsigned short* We1t  = (unsigned short*)(ws + alloc((size_t)EEXP * MMOE * HHALF * 2)); // [e][h][m]
  unsigned short* We2t  = (unsigned short*)(ws + alloc((size_t)EEXP * HHALF * MMOE * 2)); // [e][m][h]
  float*          oflF  = (float*)(ws + alloc((size_t)NTOK * MMOE * 4));
  unsigned short* oflB  = (unsigned short*)(ws + alloc((size_t)NTOK * MMOE * 2));
  unsigned short* c2sWt = (unsigned short*)(ws + alloc((size_t)MMOE * HDIM * 2));     // [2048][64]
  unsigned short* decWt = (unsigned short*)(ws + alloc((size_t)HDIM * DMODEL * 2));   // [1024][2048]

  // activations: plain convert; weights: transpose-convert to [N][K] bf16
  f32_to_bf16_kernel<<<(NTOK * DMODEL + 255) / 256, 256, 0, stream>>>(X, Xbf, NTOK * DMODEL);
  transpose_cvt_kernel<<<dim3(HDIM / 64, DMODEL / 64, 1), 256, 0, stream>>>(enc_W, encWt, DMODEL, HDIM);
  transpose_cvt_kernel<<<dim3(MMOE / 64, HDIM / 64, 1), 256, 0, stream>>>(s2c_W, s2cWt, HDIM, MMOE);
  transpose_cvt_kernel<<<dim3(HHALF / 64, MMOE / 64, EEXP), 256, 0, stream>>>(We1, We1t, MMOE, HHALF);
  transpose_cvt_kernel<<<dim3(MMOE / 64, HHALF / 64, EEXP), 256, 0, stream>>>(We2, We2t, HHALF, MMOE);
  transpose_cvt_kernel<<<dim3(HDIM / 64, MMOE / 64, 1), 256, 0, stream>>>(c2s_W, c2sWt, MMOE, HDIM);
  transpose_cvt_kernel<<<dim3(DMODEL / 64, HDIM / 64, 1), 256, 0, stream>>>(dec_W, decWt, HDIM, DMODEL);

  // 1) spikes_enc = sigmoid(X @ enc_W + enc_b)        (8192x1024 -> 8192x2048, bf16)
  gemm_bf16_big_kernel<<<dim3(HDIM / 128, NTOK / 128), 256, 0, stream>>>(
      Xbf, encWt, enc_b, nullptr, spkBF, NTOK, HDIM, DMODEL, 1);
  // 2) cont = spikes_enc @ s2c_W + s2c_b              (N=64 path) -> fp32 + bf16
  gemm_bf16_n64_kernel<<<dim3(1, NTOK / 64), 128, 0, stream>>>(
      spkBF, s2cWt, s2c_b, contF, contB, NTOK, HDIM, 0);
  // 3) router -> dense gates
  router_kernel<<<NTOK / 256, 256, 0, stream>>>(contF, rW1, rb1, rW2, rb2, gatesP);
  // 4) dense MoE experts -> out_flat
  expert_kernel<<<NTOK / 64, 128, 0, stream>>>(contB, gatesP, We1t, be1, We2t, be2, oflF, oflB);
  // 5) spikes_moe = sigmoid(out_flat @ c2s_W + c2s_b) (8192x64 -> 8192x2048, reuse spk slab)
  gemm_bf16_big_kernel<<<dim3(HDIM / 128, NTOK / 128), 256, 0, stream>>>(
      oflB, c2sWt, c2s_b, nullptr, spkBF, NTOK, HDIM, MMOE, 1);
  // 6) decoded = sigmoid(spikes_moe @ dec_W + dec_b)  -> fp32 straight into d_out
  gemm_bf16_big_kernel<<<dim3(DMODEL / 128, NTOK / 128), 256, 0, stream>>>(
      spkBF, decWt, dec_b, out, nullptr, NTOK, DMODEL, HDIM, 1);
  // 7) LayerNorm in place on d_out
  ln_kernel<<<NTOK, 256, 0, stream>>>(out, ln_g, ln_b);

  (void)in_sizes; (void)n_in; (void)out_size; (void)ws_size;
}
